// GRUEncoder_17781164605880
// MI455X (gfx1250) — matrix operand order repaired and checked by reading, not compiled, not run on hardware
//
#include <hip/hip_runtime.h>
#include <hip/hip_bf16.h>
#include <math.h>

// ---------------- problem constants ----------------
constexpr int BATCH = 256;
constexpr int SEQ   = 128;
constexpr int IN    = 1024;   // 256 + 768
constexpr int HID   = 1024;
constexpr int G3    = 3 * HID;          // 3072
constexpr int MTOT  = BATCH * SEQ;      // 32768 rows of x_proj

typedef __attribute__((ext_vector_type(16))) __bf16 v16bf;
typedef __attribute__((ext_vector_type(8)))  __bf16 v8bf;
typedef __attribute__((ext_vector_type(8)))  float  v8f;

union Frag16 { v16bf v; v8bf h[2]; };

// ---------------- conversion / init kernels ----------------

// Build concatenated bf16 input matrix X (MTOT x IN): [0,256)=base, [256,1024)=visual
__global__ void build_x_bf16(const float* __restrict__ base,
                             const float* __restrict__ vis,
                             __bf16* __restrict__ X) {
    size_t idx = (size_t)blockIdx.x * blockDim.x + threadIdx.x;
    if (idx >= (size_t)MTOT * IN) return;
    size_t m = idx >> 10;           // / 1024
    int    i = (int)(idx & 1023);
    float v = (i < 256) ? base[m * 256 + i] : vis[m * 768 + (i - 256)];
    X[idx] = (__bf16)v;
}

__global__ void cvt_f32_bf16(const float* __restrict__ src,
                             __bf16* __restrict__ dst, size_t n) {
    size_t idx = (size_t)blockIdx.x * blockDim.x + threadIdx.x;
    if (idx < n) dst[idx] = (__bf16)src[idx];
}

__global__ void init_h(float* __restrict__ hf, __bf16* __restrict__ hb) {
    size_t idx = (size_t)blockIdx.x * blockDim.x + threadIdx.x;
    if (idx < (size_t)BATCH * HID) { hf[idx] = 0.0f; hb[idx] = (__bf16)0.0f; }
}

// ---------------- Phase 1: x_proj = X * W_ih^T + b_ih  (WMMA bf16) ----------------
// Register-blocked: one wave computes a 32x64 region of P (2 m-tiles x 4 n-tiles),
// 8 WMMAs per K-step of 32 with only 6 fragment loads (A reused across n, B across m).
__global__ __launch_bounds__(256)
void xproj_wmma(const __bf16* __restrict__ X,    // (MTOT, IN)
                const __bf16* __restrict__ W,    // (G3, IN) row-major -> B fragment contiguous
                const float*  __restrict__ bias, // (G3)
                float* __restrict__ P)           // (MTOT, G3)
{
    constexpr int MT = 2, NT = 4;
    const int nsuper = (G3 / 16) / NT;              // 48 super-tiles along N
    int wave = blockIdx.x * (blockDim.x >> 5) + (threadIdx.x >> 5);
    int lane = threadIdx.x & 31;
    int l16  = lane & 15;
    int hi   = lane >> 4;
    int ms   = wave / nsuper;                       // 0..1023
    int ns   = wave - ms * nsuper;                  // 0..47
    int m0 = ms * (16 * MT);
    int n0 = ns * (16 * NT);

    // A lane row: 16 contiguous K at offset hi*16
    const __bf16* arow[MT];
    #pragma unroll
    for (int i = 0; i < MT; ++i)
        arow[i] = X + (size_t)(m0 + i * 16 + l16) * IN + hi * 16;
    // B lane col: 16 contiguous K at offset hi*16
    const __bf16* brow[NT];
    #pragma unroll
    for (int j = 0; j < NT; ++j)
        brow[j] = W + (size_t)(n0 + j * 16 + l16) * IN + hi * 16;

    v8f c[MT][NT] = {};
    for (int kk = 0; kk < IN; kk += 32) {
        Frag16 a[MT], b[NT];
        #pragma unroll
        for (int i = 0; i < MT; ++i) {
            a[i].h[0] = *(const v8bf*)(arow[i] + kk);
            a[i].h[1] = *(const v8bf*)(arow[i] + kk + 8);
        }
        #pragma unroll
        for (int j = 0; j < NT; ++j) {
            b[j].h[0] = *(const v8bf*)(brow[j] + kk);
            b[j].h[1] = *(const v8bf*)(brow[j] + kk + 8);
        }
        #pragma unroll
        for (int i = 0; i < MT; ++i)
            #pragma unroll
            for (int j = 0; j < NT; ++j)
                c[i][j] = __builtin_amdgcn_wmma_f32_16x16x32_bf16(
                    false, a[i].v, false, b[j].v, (short)0, c[i][j], false, false);
    }

    #pragma unroll
    for (int j = 0; j < NT; ++j) {
        float bv = bias[n0 + j * 16 + l16];
        #pragma unroll
        for (int i = 0; i < MT; ++i) {
            float* prow = P + (size_t)(m0 + i * 16 + hi * 8) * G3 + n0 + j * 16 + l16;
            #pragma unroll
            for (int q = 0; q < 8; ++q) prow[(size_t)q * G3] = c[i][j][q] + bv;
        }
    }
}

// ---------------- Phase 2: one GRU step (WMMA bf16 + fused gates) ----------------
// Wave (ms, ct): 2 m-tiles x {r,z,n} tiles at column ct*16. The three W_hh fragments
// are reused across both m-tiles: 6 WMMAs per 5 fragment loads.
__global__ __launch_bounds__(256)
void gru_step(const __bf16* __restrict__ Hb,   // (BATCH, HID) h_{t-1} bf16
              const float*  __restrict__ Hf,   // (BATCH, HID) h_{t-1} f32
              const __bf16* __restrict__ Whh,  // (G3, HID)
              const float*  __restrict__ bhh,  // (G3)
              const float*  __restrict__ P,    // x_proj (MTOT, G3)
              float*  __restrict__ HfOut,
              __bf16* __restrict__ HbOut,
              int step)
{
    constexpr int MT = 2;
    const int ctiles = HID / 16;                   // 64
    int wave = blockIdx.x * (blockDim.x >> 5) + (threadIdx.x >> 5);
    int lane = threadIdx.x & 31;
    int l16  = lane & 15;
    int hi   = lane >> 4;
    int ct   = wave & (ctiles - 1);
    int ms   = wave >> 6;                          // 0..7 (BATCH/32)
    int m0 = ms * (16 * MT);
    int col = ct * 16 + l16;                       // h column this lane owns

    const __bf16* arow[MT];
    #pragma unroll
    for (int i = 0; i < MT; ++i)
        arow[i] = Hb + (size_t)(m0 + i * 16 + l16) * HID + hi * 16;
    const __bf16* browR = Whh + (size_t)col * HID + hi * 16;
    const __bf16* browZ = browR + (size_t)HID * HID;
    const __bf16* browN = browR + (size_t)2 * HID * HID;

    v8f cr[MT] = {}, cz[MT] = {}, cn[MT] = {};
    for (int kk = 0; kk < HID; kk += 32) {
        Frag16 a[MT], br, bz, bn;
        #pragma unroll
        for (int i = 0; i < MT; ++i) {
            a[i].h[0] = *(const v8bf*)(arow[i] + kk);
            a[i].h[1] = *(const v8bf*)(arow[i] + kk + 8);
        }
        br.h[0] = *(const v8bf*)(browR + kk);
        br.h[1] = *(const v8bf*)(browR + kk + 8);
        bz.h[0] = *(const v8bf*)(browZ + kk);
        bz.h[1] = *(const v8bf*)(browZ + kk + 8);
        bn.h[0] = *(const v8bf*)(browN + kk);
        bn.h[1] = *(const v8bf*)(browN + kk + 8);
        #pragma unroll
        for (int i = 0; i < MT; ++i) {
            cr[i] = __builtin_amdgcn_wmma_f32_16x16x32_bf16(false, a[i].v, false, br.v, (short)0, cr[i], false, false);
            cz[i] = __builtin_amdgcn_wmma_f32_16x16x32_bf16(false, a[i].v, false, bz.v, (short)0, cz[i], false, false);
            cn[i] = __builtin_amdgcn_wmma_f32_16x16x32_bf16(false, a[i].v, false, bn.v, (short)0, cn[i], false, false);
        }
    }

    float br_ = bhh[col];
    float bz_ = bhh[col + HID];
    float bn_ = bhh[col + 2 * HID];

    #pragma unroll
    for (int i = 0; i < MT; ++i) {
        #pragma unroll
        for (int j = 0; j < 8; ++j) {
            int b = m0 + i * 16 + hi * 8 + j;              // batch row
            size_t prow = ((size_t)b * SEQ + step) * G3 + col;
            float xr = P[prow];
            float xz = P[prow + HID];
            float xn = P[prow + 2 * HID];
            float hr = cr[i][j] + br_;
            float hz = cz[i][j] + bz_;
            float hn = cn[i][j] + bn_;
            float r = 1.0f / (1.0f + __expf(-(xr + hr)));
            float z = 1.0f / (1.0f + __expf(-(xz + hz)));
            float n = tanhf(xn + r * hn);
            float hold = Hf[(size_t)b * HID + col];
            float hnew = (1.0f - z) * n + z * hold;
            HfOut[(size_t)b * HID + col] = hnew;
            HbOut[(size_t)b * HID + col] = (__bf16)hnew;
        }
    }
}

// ---------------- host side ----------------
extern "C" void kernel_launch(void* const* d_in, const int* in_sizes, int n_in,
                              void* d_out, int out_size, void* d_ws, size_t ws_size,
                              hipStream_t stream) {
    const float* base = (const float*)d_in[0];   // (256,128,256)
    const float* vis  = (const float*)d_in[1];   // (256,128,768)
    const float* Wih  = (const float*)d_in[2];   // (3072,1024)
    const float* Whh  = (const float*)d_in[3];   // (3072,1024)
    const float* bih  = (const float*)d_in[4];   // (3072)
    const float* bhh  = (const float*)d_in[5];   // (3072)
    float* out = (float*)d_out;

    char* ws = (char*)d_ws;
    size_t off = 0;
    auto carve = [&](size_t bytes) -> void* {
        void* p = ws + off;
        off += (bytes + 255) & ~(size_t)255;
        return p;
    };

    float*  P     = (float*) carve((size_t)MTOT * G3 * sizeof(float));   // 402 MB
    __bf16* Xb    = (__bf16*)carve((size_t)MTOT * IN * sizeof(__bf16));  // 64 MB
    __bf16* WihB  = (__bf16*)carve((size_t)G3 * IN * sizeof(__bf16));    // 6 MB
    __bf16* WhhB  = (__bf16*)carve((size_t)G3 * HID * sizeof(__bf16));   // 6 MB
    float*  Hf[2];
    __bf16* Hb[2];
    Hf[0] = (float*) carve((size_t)BATCH * HID * sizeof(float));
    Hf[1] = (float*) carve((size_t)BATCH * HID * sizeof(float));
    Hb[0] = (__bf16*)carve((size_t)BATCH * HID * sizeof(__bf16));
    Hb[1] = (__bf16*)carve((size_t)BATCH * HID * sizeof(__bf16));

    // --- conversions ---
    {
        size_t n = (size_t)MTOT * IN;
        build_x_bf16<<<(unsigned)((n + 255) / 256), 256, 0, stream>>>(base, vis, Xb);
    }
    {
        size_t n = (size_t)G3 * IN;
        cvt_f32_bf16<<<(unsigned)((n + 255) / 256), 256, 0, stream>>>(Wih, WihB, n);
    }
    {
        size_t n = (size_t)G3 * HID;
        cvt_f32_bf16<<<(unsigned)((n + 255) / 256), 256, 0, stream>>>(Whh, WhhB, n);
    }
    {
        size_t n = (size_t)BATCH * HID;
        init_h<<<(unsigned)((n + 255) / 256), 256, 0, stream>>>(Hf[0], Hb[0]);
    }

    // --- Phase 1: x_proj GEMM (32x64 per wave) ---
    {
        int waves  = (MTOT / 32) * (G3 / 64);      // 1024 * 48 = 49152
        int blocks = waves / 8;                    // 8 waves (256 thr) per block
        xproj_wmma<<<blocks, 256, 0, stream>>>(Xb, WihB, bih, P);
    }

    // --- Phase 2: sequential GRU scan (double-buffered h) ---
    {
        int waves  = (BATCH / 32) * (HID / 16);    // 8 * 64 = 512
        int blocks = waves / 8;                    // 64
        for (int k = 0; k < SEQ; ++k) {
            int src = k & 1, dst = (k + 1) & 1;
            gru_step<<<blocks, 256, 0, stream>>>(Hb[src], Hf[src], WhhB, bhh, P,
                                                 Hf[dst], Hb[dst], k);
        }
    }

    // SEQ is even -> final h lands in buffer 0
    hipMemcpyAsync(out, Hf[0], (size_t)BATCH * HID * sizeof(float),
                   hipMemcpyDeviceToDevice, stream);
}
